// MIL_Feature_Attention_MultiBag_29051158790278
// MI455X (gfx1250) — matrix operand order repaired and checked by reading, not compiled, not run on hardware
//
#include <hip/hip_runtime.h>
#include <hip/hip_bf16.h>
#include <math.h>

typedef __attribute__((ext_vector_type(16))) _Float16 v16h;
typedef __attribute__((ext_vector_type(8)))  float    v8f;

#define B_  128
#define T_  500
#define M_  512
#define L_  128

// Packed weight layout (half index):
//   p = (((kc*2 + mat)*8 + lt)*32 + lane)*16 + h
// where kc = K-chunk (0..15, 32 K each), mat = 0:Wv 1:Wu, lt = L-tile (0..7),
// lane = wave lane, h = element in v16h.  This is this code's contiguous-K 16-bit
// B-matrix (32x16) per-lane layout: column N = lt*16 + (lane&15),
// K = kc*32 + 16*(lane>>4) + h.
#define PK_HALVES (16 * 2 * 8 * 32 * 16)   // 131072 halves = 256 KB

__global__ __launch_bounds__(256) void prep_weights_kernel(
    const float* __restrict__ Wv, const float* __restrict__ Wu,
    _Float16* __restrict__ pk)
{
    int o    = blockIdx.x * 256 + threadIdx.x;      // 512 blocks * 256 = 131072
    int h    = o & 15;
    int lane = (o >> 4) & 31;
    int lt   = (o >> 9) & 7;
    int mat  = (o >> 12) & 1;
    int kc   = (o >> 13) & 15;
    int k    = kc * 32 + ((lane >> 4) << 4) + h;    // K index in [0,512)
    int col  = lt * 16 + (lane & 15);               // L index in [0,128)
    const float* W = mat ? Wu : Wv;
    pk[o] = (_Float16)W[k * L_ + col];
}

// One block = 8 waves, each wave owns a 16-row t-tile (block covers 128 rows).
// Weights staged through 64KB LDS in 4 K-stages; H read once from HBM.
__global__ __launch_bounds__(256) void gated_logits_kernel(
    const float* __restrict__ H, const _Float16* __restrict__ pk,
    const float* __restrict__ bv, const float* __restrict__ bu,
    const float* __restrict__ Wa, const float* __restrict__ ba,
    float* __restrict__ logits)
{
    __shared__ _Float16 sw[4 * 2 * 8 * 32 * 16];    // 32768 halves = 64 KB

    const int tid  = threadIdx.x;
    const int b    = blockIdx.x >> 2;
    const int tblk = blockIdx.x & 3;
    const int wave = tid >> 5;
    const int lane = tid & 31;
    const int g    = lane >> 4;       // lane group (0/1)
    const int n    = lane & 15;       // lane within group
    const int t0   = tblk * 128 + wave * 16;

    // A-matrix: lane holds row (l & 15); clamp padded rows (t >= 500) safely.
    int trow = t0 + n;
    int hrow = trow < T_ ? trow : (T_ - 1);
    const float* hptr = H + ((size_t)b * T_ + hrow) * M_;

    v8f accV[8], accU[8];
    const v8f vzero = {0.f, 0.f, 0.f, 0.f, 0.f, 0.f, 0.f, 0.f};
#pragma unroll
    for (int i = 0; i < 8; ++i) { accV[i] = vzero; accU[i] = vzero; }

    const float4* pk4 = (const float4*)pk;
    float4*       sw4 = (float4*)sw;

    for (int ks = 0; ks < 4; ++ks) {
        __syncthreads();                            // waves done with prev stage
#pragma unroll
        for (int i = 0; i < 16; ++i)                // 4096 float4 = 64 KB
            sw4[i * 256 + tid] = pk4[ks * 4096 + i * 256 + tid];
        __syncthreads();

#pragma unroll
        for (int kc2 = 0; kc2 < 4; ++kc2) {
            const int k0 = (ks * 4 + kc2) * 32;
            // 16-bit A (16x32) per-lane order used here, contiguous K (matches packed B):
            //   K = 16*g + h
            float4 f0 = *(const float4*)(hptr + k0 + 16 * g);
            float4 f1 = *(const float4*)(hptr + k0 + 16 * g + 4);
            float4 f2 = *(const float4*)(hptr + k0 + 16 * g + 8);
            float4 f3 = *(const float4*)(hptr + k0 + 16 * g + 12);
            v16h a;
            a[0]  = (_Float16)f0.x; a[1]  = (_Float16)f0.y;
            a[2]  = (_Float16)f0.z; a[3]  = (_Float16)f0.w;
            a[4]  = (_Float16)f1.x; a[5]  = (_Float16)f1.y;
            a[6]  = (_Float16)f1.z; a[7]  = (_Float16)f1.w;
            a[8]  = (_Float16)f2.x; a[9]  = (_Float16)f2.y;
            a[10] = (_Float16)f2.z; a[11] = (_Float16)f2.w;
            a[12] = (_Float16)f3.x; a[13] = (_Float16)f3.y;
            a[14] = (_Float16)f3.z; a[15] = (_Float16)f3.w;

#pragma unroll
            for (int lt = 0; lt < 8; ++lt) {
                const v16h bvm = *(const v16h*)&sw[(((kc2 * 2 + 0) * 8 + lt) * 32 + lane) * 16];
                accV[lt] = __builtin_amdgcn_wmma_f32_16x16x32_f16(
                    false, a, false, bvm, (short)0, accV[lt], false, false);
                const v16h bum = *(const v16h*)&sw[(((kc2 * 2 + 1) * 8 + lt) * 32 + lane) * 16];
                accU[lt] = __builtin_amdgcn_wmma_f32_16x16x32_f16(
                    false, a, false, bum, (short)0, accU[lt], false, false);
            }
        }
    }

    // Epilogue: gate + dot with Wa. C/D layout: VGPR r, lane -> row t0+r+8g,
    // col = lt*16 + n.  Reduce over the 16 lanes of each group.
    float part[8] = {0.f, 0.f, 0.f, 0.f, 0.f, 0.f, 0.f, 0.f};
#pragma unroll
    for (int lt = 0; lt < 8; ++lt) {
        const int col = lt * 16 + n;
        const float bvv = bv[col], buu = bu[col], wav = Wa[col];
#pragma unroll
        for (int r = 0; r < 8; ++r) {
            float v = tanhf(accV[lt][r] + bvv);
            float u = accU[lt][r] + buu;
            float s = 1.0f / (1.0f + expf(-u));
            part[r] += v * s * wav;
        }
    }
    const float bav = ba[0];
#pragma unroll
    for (int r = 0; r < 8; ++r) {
        float x = part[r];
        x += __shfl_xor(x, 1);
        x += __shfl_xor(x, 2);
        x += __shfl_xor(x, 4);
        x += __shfl_xor(x, 8);
        int t = t0 + r + 8 * g;
        if (n == 0 && t < T_)
            logits[(size_t)b * T_ + t] = x + bav;   // this IS output A_t
    }
}

// One block per batch: softmax over T, attention pooling (GEMV over H),
// final 512x2 projection.
__global__ __launch_bounds__(128) void softmax_pool_kernel(
    const float* __restrict__ H, const float* __restrict__ logits,
    const float* __restrict__ Wc, const float* __restrict__ bc,
    float* __restrict__ out, float* __restrict__ a_sm)
{
    __shared__ float red[128];
    __shared__ float ew[T_];
    const int b = blockIdx.x, tid = threadIdx.x;
    const float* lg = logits + (size_t)b * T_;

    float mx = -1e30f;
    for (int t = tid; t < T_; t += 128) mx = fmaxf(mx, lg[t]);
    red[tid] = mx; __syncthreads();
    for (int s = 64; s > 0; s >>= 1) {
        if (tid < s) red[tid] = fmaxf(red[tid], red[tid + s]);
        __syncthreads();
    }
    mx = red[0]; __syncthreads();

    float sum = 0.f;
    for (int t = tid; t < T_; t += 128) { float e = expf(lg[t] - mx); ew[t] = e; sum += e; }
    red[tid] = sum; __syncthreads();
    for (int s = 64; s > 0; s >>= 1) {
        if (tid < s) red[tid] += red[tid + s];
        __syncthreads();
    }
    const float inv = 1.0f / red[0];

    for (int t = tid; t < T_; t += 128) a_sm[(size_t)b * T_ + t] = ew[t] * inv;
    __syncthreads();

    // Mpool[m] = sum_t softmax[t] * H[b,t,m]; thread owns float4 column m0..m0+3
    const float4* Hb = (const float4*)(H + (size_t)b * T_ * M_);
    float4 acc = make_float4(0.f, 0.f, 0.f, 0.f);
    for (int t = 0; t < T_; ++t) {
        const float w = ew[t] * inv;
        const float4 hv = Hb[(size_t)t * 128 + tid];
        acc.x += w * hv.x; acc.y += w * hv.y; acc.z += w * hv.z; acc.w += w * hv.w;
    }
    const int m0 = 4 * tid;
    float p0 = acc.x * Wc[(m0 + 0) * 2 + 0] + acc.y * Wc[(m0 + 1) * 2 + 0] +
               acc.z * Wc[(m0 + 2) * 2 + 0] + acc.w * Wc[(m0 + 3) * 2 + 0];
    float p1 = acc.x * Wc[(m0 + 0) * 2 + 1] + acc.y * Wc[(m0 + 1) * 2 + 1] +
               acc.z * Wc[(m0 + 2) * 2 + 1] + acc.w * Wc[(m0 + 3) * 2 + 1];
    __syncthreads();
    red[tid] = p0; __syncthreads();
    for (int s = 64; s > 0; s >>= 1) {
        if (tid < s) red[tid] += red[tid + s];
        __syncthreads();
    }
    const float o0 = red[0];
    __syncthreads();
    red[tid] = p1; __syncthreads();
    for (int s = 64; s > 0; s >>= 1) {
        if (tid < s) red[tid] += red[tid + s];
        __syncthreads();
    }
    if (tid == 0) {
        out[b * 2 + 0] = o0 + bc[0];
        out[b * 2 + 1] = red[0] + bc[1];
    }
}

extern "C" void kernel_launch(void* const* d_in, const int* in_sizes, int n_in,
                              void* d_out, int out_size, void* d_ws, size_t ws_size,
                              hipStream_t stream)
{
    (void)in_sizes; (void)n_in; (void)out_size; (void)ws_size;
    const float* H  = (const float*)d_in[0];
    const float* Wv = (const float*)d_in[1];
    const float* bv = (const float*)d_in[2];
    const float* Wu = (const float*)d_in[3];
    const float* bu = (const float*)d_in[4];
    const float* Wa = (const float*)d_in[5];
    const float* ba = (const float*)d_in[6];
    const float* Wc = (const float*)d_in[7];
    const float* bc = (const float*)d_in[8];

    float* out  = (float*)d_out;          // [B,2]
    float* a_sm = out + B_ * 2;           // [B,T]   softmax weights
    float* a_t  = a_sm + B_ * T_;         // [B,1,T] raw logits (output 3)

    _Float16* pk = (_Float16*)d_ws;       // 256 KB packed f16 weights

    prep_weights_kernel<<<PK_HALVES / 256, 256, 0, stream>>>(Wv, Wu, pk);
    gated_logits_kernel<<<B_ * 4, 256, 0, stream>>>(H, pk, bv, bu, Wa, ba, a_t);
    softmax_pool_kernel<<<B_, 128, 0, stream>>>(H, a_t, Wc, bc, out, a_sm);
}
